// ResidualBlock_63926293233953
// MI455X (gfx1250) — matrix operand order repaired and checked by reading, not compiled, not run on hardware
//
#include <hip/hip_runtime.h>

typedef __attribute__((ext_vector_type(16))) __bf16 bf16x16;
typedef __attribute__((ext_vector_type(8)))  float  f32x8;
typedef __attribute__((ext_vector_type(4)))  unsigned int u32x4;

#define LDSS 48   // LDS row stride (bf16 elems) for 32-wide K tiles; 96B = 6*16B

union FragU { bf16x16 v; u32x4 q[2]; };

// Generic-pointer low 32 bits == LDS byte offset (ISA 10.2: LDS aperture truncates addr[31:0]).
__device__ inline unsigned lds_addr_of(const void* p) {
  return (unsigned)(unsigned long long)p;
}
// CDNA5 async global->LDS copy, 16B per lane, tracked by ASYNCcnt.
__device__ inline void async_load_b128(unsigned lds, const void* gaddr) {
  asm volatile("global_load_async_to_lds_b128 %0, %1, off" :: "v"(lds), "v"(gaddr) : "memory");
}
__device__ inline void wait_asynccnt0() {
  asm volatile("s_wait_asynccnt 0" ::: "memory");
}

// ---------------------------------------------------------------------------
// Wave-level 32x64 (M x N) x 32 (K) WMMA step on LDS tiles.
// As: [128][LDSS] bf16 (rows=M, cols=K in [0,32))
// Bs: [128][LDSS] bf16 (rows=N, cols=K in [0,32))  -- B^T layout in LDS
// ---------------------------------------------------------------------------
__device__ inline void wave_mma_8(const __bf16* As, const __bf16* Bs,
                                  f32x8 acc[2][4], int wm, int wn, int lane) {
  const int lh = lane >> 4;
  const int lm = lane & 15;
  FragU a[2], b[4];
#pragma unroll
  for (int i = 0; i < 2; i++) {
    const __bf16* p = As + (wm * 32 + i * 16 + lm) * LDSS;
    // A 16x32 bf16 layout: lanes 0-15: K {0..7,16..23}; lanes 16-31: K {8..15,24..31}
    a[i].q[0] = *(const u32x4*)(p + lh * 8);
    a[i].q[1] = *(const u32x4*)(p + 16 + lh * 8);
  }
#pragma unroll
  for (int j = 0; j < 4; j++) {
    // B 32x16 bf16 layout: col = lane&15; lanes 0-15: K {0..7,16..23}; lanes 16-31: K {8..15,24..31}
    const __bf16* p = Bs + (wn * 64 + j * 16 + lm) * LDSS;
    b[j].q[0] = *(const u32x4*)(p + lh * 8);
    b[j].q[1] = *(const u32x4*)(p + 16 + lh * 8);
  }
#pragma unroll
  for (int i = 0; i < 2; i++)
#pragma unroll
    for (int j = 0; j < 4; j++)
      acc[i][j] = __builtin_amdgcn_wmma_f32_16x16x32_bf16(
          false, a[i].v, false, b[j].v, (short)0, acc[i][j], false, false);
}

// ---------------------------------------------------------------------------
// Generic batched GEMM:  C[z] = op( A[z] (M x K) * B[z] (K x N) [+bias] [+resid] )
// TRANSB: B stored as [N x K] row-major (compute A * B^T).
// Ping-pong LDS, async global->LDS staging, one barrier per 32-deep K step.
// Block tile 128x128. grid = (N/128, M/128, batches), 256 threads.
// ---------------------------------------------------------------------------
template <bool TRANSB, bool BIAS, bool RELU, bool RESID, bool TRIL, bool OUTBF16>
__global__ __launch_bounds__(256) void gemm_kernel(
    const __bf16* __restrict__ A, long lda, long sA,
    const __bf16* __restrict__ B, long ldb, long sB,
    const float* __restrict__ bias,
    void* __restrict__ Cout, long ldc, long sC,
    const float* __restrict__ resid, long sR, int KK) {
  __shared__ __align__(16) __bf16 As[2][128 * LDSS];
  __shared__ __align__(16) __bf16 Bs[2][128 * LDSS];
  const int tid = threadIdx.x;
  const int bM = blockIdx.y * 128, bN = blockIdx.x * 128;
  const int z = blockIdx.z;
  A += (long)z * sA + (long)bM * lda;
  B += (long)z * sB;
  const int lane = tid & 31, w = tid >> 5, wm = w >> 1, wn = w & 1;
  f32x8 acc[2][4];
  const f32x8 zero = {0, 0, 0, 0, 0, 0, 0, 0};
#pragma unroll
  for (int i = 0; i < 2; i++)
#pragma unroll
    for (int j = 0; j < 4; j++) acc[i][j] = zero;

  int nsteps = KK / 32;
  // strict-lower-tri output: blocks fully above the diagonal are all zeros
  if (TRIL && bM < bN) nsteps = 0;

  const int rowA = tid >> 1, halfA = (tid & 1) * 16;  // contiguous 32B per thread
  const int kB = tid >> 4, n8 = (tid & 15) * 8;       // transpose path indices

  auto stageA = [&](int k0, int buf) {
    const __bf16* g = A + (long)rowA * lda + k0 + halfA;
    unsigned l = lds_addr_of(&As[buf][rowA * LDSS + halfA]);
    async_load_b128(l, g);
    async_load_b128(l + 16, g + 8);
  };
  auto stageBtrans = [&](int k0, int buf) {
    const __bf16* g = B + (long)(bN + rowA) * ldb + k0 + halfA;
    unsigned l = lds_addr_of(&Bs[buf][rowA * LDSS + halfA]);
    async_load_b128(l, g);
    async_load_b128(l + 16, g + 8);
  };

  if (nsteps > 0) {  // prologue: stage tile 0 into buffer 0
    stageA(0, 0);
    if (TRANSB) {
      stageBtrans(0, 0);
    } else {
#pragma unroll
      for (int kk2 = 0; kk2 < 2; kk2++) {
        int kr = kB + kk2 * 16;
        union { u32x4 u; __bf16 h[8]; } t;
        t.u = *(const u32x4*)(B + (long)kr * ldb + bN + n8);
#pragma unroll
        for (int j = 0; j < 8; j++) Bs[0][(n8 + j) * LDSS + kr] = t.h[j];
      }
    }
    wait_asynccnt0();
    __syncthreads();
  }

  for (int s = 0; s < nsteps; s++) {
    const int p = s & 1;
    const bool hasNext = (s + 1 < nsteps);
    union { u32x4 u; __bf16 h[8]; } t0, t1;
    if (hasNext) {  // issue next tile's loads before compute (latency overlap)
      int k0n = (s + 1) * 32;
      stageA(k0n, 1 - p);
      if (TRANSB) {
        stageBtrans(k0n, 1 - p);
      } else {
        t0.u = *(const u32x4*)(B + (long)(k0n + kB) * ldb + bN + n8);
        t1.u = *(const u32x4*)(B + (long)(k0n + kB + 16) * ldb + bN + n8);
      }
    }
    wave_mma_8(&As[p][0], &Bs[p][0], acc, wm, wn, lane);
    if (hasNext && !TRANSB) {
#pragma unroll
      for (int j = 0; j < 8; j++) Bs[1 - p][(n8 + j) * LDSS + kB] = t0.h[j];
#pragma unroll
      for (int j = 0; j < 8; j++) Bs[1 - p][(n8 + j) * LDSS + kB + 16] = t1.h[j];
    }
    if (hasNext) {
      wait_asynccnt0();
      __syncthreads();
    }
  }

  const int lh = lane >> 4, lm = lane & 15;
#pragma unroll
  for (int i = 0; i < 2; i++)
#pragma unroll
    for (int j = 0; j < 4; j++) {
      int n = bN + wn * 64 + j * 16 + lm;
      float bv = BIAS ? bias[n] : 0.f;
#pragma unroll
      for (int r = 0; r < 8; r++) {
        int m = bM + wm * 32 + i * 16 + r + 8 * lh;
        float val = acc[i][j][r] + bv;
        if (RESID) val += resid[(long)z * sR + (long)m * ldc + n];
        if (RELU) val = fmaxf(val, 0.f);
        if (TRIL) val = (m > n) ? val : 0.f;
        long off = (long)z * sC + (long)m * ldc + n;
        if (OUTBF16) ((__bf16*)Cout)[off] = (__bf16)val;
        else         ((float*)Cout)[off] = val;
      }
    }
}

// ---------------------------------------------------------------------------
// Blocked forward substitution step for (I + K) u = v, K strictly lower.
//   W = v[I, cols] - K[I, 0:I0] @ u[0:I0, cols]      (WMMA)
//   then 128-step unit-diagonal solve in fp32 LDS, write u and lr*u (bf16).
// grid = (C/128 column chunks, N batches), 256 threads.
// ---------------------------------------------------------------------------
__global__ __launch_bounds__(256) void solve_step_kernel(
    const __bf16* __restrict__ Kmat,  // [N, T, T] bf16, strictly lower
    __bf16* __restrict__ u, __bf16* __restrict__ ulr,
    const __bf16* __restrict__ v, const float* __restrict__ lr, int step) {
  const int T = 2048, C = 1024;
  __shared__ __align__(16) __bf16 As[128 * LDSS];
  __shared__ __align__(16) __bf16 Bs[128 * LDSS];
  __shared__ float Ws[128][132];
  __shared__ __align__(16) __bf16 Kd[128][136];
  const int tid = threadIdx.x;
  const int n = blockIdx.y;
  const int x0 = blockIdx.x * 128;
  const int I0 = step * 128;
  const __bf16* Arow = Kmat + (long)n * T * T + (long)I0 * T;  // [128, T]
  const __bf16* Bu = u + (long)n * T * C + x0;                  // [KK, 128], ldb=C
  const int lane = tid & 31, w = tid >> 5, wm = w >> 1, wn = w & 1;
  f32x8 acc[2][4];
  const f32x8 zero = {0, 0, 0, 0, 0, 0, 0, 0};
#pragma unroll
  for (int i = 0; i < 2; i++)
#pragma unroll
    for (int j = 0; j < 4; j++) acc[i][j] = zero;

  const int KK = I0;
  for (int k0 = 0; k0 < KK; k0 += 32) {
    {
      int row = tid >> 1, half = (tid & 1) * 16;
      const u32x4* s = (const u32x4*)(Arow + (long)row * T + k0 + half);
      u32x4 v0 = s[0], v1 = s[1];
      u32x4* d = (u32x4*)(As + row * LDSS + half);
      d[0] = v0; d[1] = v1;
    }
    {
      int k = tid >> 4, n8 = (tid & 15) * 8;
#pragma unroll
      for (int kk2 = 0; kk2 < 2; kk2++) {
        int kr = k + kk2 * 16;
        union { u32x4 uu; __bf16 h[8]; } t;
        t.uu = *(const u32x4*)(Bu + (long)(k0 + kr) * C + n8);
#pragma unroll
        for (int j = 0; j < 8; j++) Bs[(n8 + j) * LDSS + kr] = t.h[j];
      }
    }
    __syncthreads();
    wave_mma_8(As, Bs, acc, wm, wn, lane);
    __syncthreads();
  }

  // spill acc (= K@u partial sums) into Ws
  const int lh = lane >> 4, lm = lane & 15;
#pragma unroll
  for (int i = 0; i < 2; i++)
#pragma unroll
    for (int j = 0; j < 4; j++) {
      int c = wn * 64 + j * 16 + lm;
#pragma unroll
      for (int r = 0; r < 8; r++) {
        int m = wm * 32 + i * 16 + r + 8 * lh;
        Ws[m][c] = acc[i][j][r];
      }
    }
  __syncthreads();

  // W = v - K@u ; load diagonal block of K
  for (int idx = tid; idx < 16384; idx += 256) {
    int t = idx >> 7, c = idx & 127;
    float vv = (float)v[(long)n * T * C + (long)(I0 + t) * C + x0 + c];
    Ws[t][c] = vv - Ws[t][c];
  }
  {
    int row = tid >> 1, half = (tid & 1) * 64;
    const u32x4* s = (const u32x4*)(Kmat + (long)n * T * T + (long)(I0 + row) * T + I0 + half);
    u32x4* d = (u32x4*)(&Kd[row][half]);
#pragma unroll
    for (int j = 0; j < 8; j++) d[j] = s[j];
  }
  __syncthreads();

  // sequential unit-diagonal solve within block (rank-1 updates)
  for (int t = 0; t < 127; t++) {
    int c = tid & 127;
    float ut = Ws[t][c];
    for (int s = t + 1 + (tid >> 7); s < 128; s += 2)
      Ws[s][c] -= (float)Kd[s][t] * ut;
    __syncthreads();
  }

  for (int idx = tid; idx < 16384; idx += 256) {
    int t = idx >> 7, c = idx & 127;
    long row = (long)n * T + I0 + t;
    float uv = Ws[t][c];
    u[row * C + x0 + c] = (__bf16)uv;
    ulr[row * C + x0 + c] = (__bf16)(lr[row] * uv);
  }
}

// ---------------------------------------------------------------------------
// LayerNorm (one row / block, C=1024, 256 threads), bf16 out; optional lr dot.
// ---------------------------------------------------------------------------
__device__ inline float block_reduce(float val, float* red, int tid) {
  red[tid] = val;
  __syncthreads();
  for (int off = 128; off > 0; off >>= 1) {
    if (tid < off) red[tid] += red[tid + off];
    __syncthreads();
  }
  float r = red[0];
  __syncthreads();
  return r;
}

template <bool DO_LR>
__global__ __launch_bounds__(256) void ln_kernel(
    const float* __restrict__ x, const float* __restrict__ g,
    const float* __restrict__ b, const float* __restrict__ lrw,
    const float* __restrict__ lrb, __bf16* __restrict__ out,
    float* __restrict__ lrout) {
  __shared__ float red[256];
  const int tid = threadIdx.x;
  const long row = blockIdx.x;
  const float* xr = x + row * 1024;
  float v[4], s = 0.f;
#pragma unroll
  for (int it = 0; it < 4; it++) { v[it] = xr[tid + it * 256]; s += v[it]; }
  float m = block_reduce(s, red, tid) * (1.f / 1024.f);
  s = 0.f;
#pragma unroll
  for (int it = 0; it < 4; it++) { float d = v[it] - m; s += d * d; }
  float var = block_reduce(s, red, tid) * (1.f / 1024.f);
  float inv = rsqrtf(var + 1e-5f);
  float dot = 0.f;
#pragma unroll
  for (int it = 0; it < 4; it++) {
    int idx = tid + it * 256;
    float h = (v[it] - m) * inv * g[idx] + b[idx];
    out[row * 1024 + idx] = (__bf16)h;
    if (DO_LR) dot += h * lrw[idx];
  }
  if (DO_LR) {
    float d = block_reduce(dot, red, tid);
    if (tid == 0) lrout[row] = d + lrb[0];
  }
}

// Normalize each row of k (bf16 in-place), C=1024.
__global__ __launch_bounds__(256) void knorm_kernel(__bf16* __restrict__ k) {
  __shared__ float red[256];
  const int tid = threadIdx.x;
  const long row = blockIdx.x;
  float v[4], s = 0.f;
#pragma unroll
  for (int it = 0; it < 4; it++) {
    v[it] = (float)k[row * 1024 + tid + it * 256];
    s += v[it] * v[it];
  }
  float inv = rsqrtf(block_reduce(s, red, tid));
#pragma unroll
  for (int it = 0; it < 4; it++)
    k[row * 1024 + tid + it * 256] = (__bf16)(v[it] * inv);
}

__global__ __launch_bounds__(256) void cvt_kernel(const float* __restrict__ in,
                                                  __bf16* __restrict__ out, long n) {
  long stride = (long)gridDim.x * 256;
  for (long i = (long)blockIdx.x * 256 + threadIdx.x; i < n; i += stride)
    out[i] = (__bf16)in[i];
}

// ---------------------------------------------------------------------------
extern "C" void kernel_launch(void* const* d_in, const int* in_sizes, int n_in,
                              void* d_out, int out_size, void* d_ws, size_t ws_size,
                              hipStream_t stream) {
  const long Nb = 8, T = 2048, C = 1024, NT = Nb * T;
  const float* x      = (const float*)d_in[0];
  const float* dn_g   = (const float*)d_in[1];
  const float* dn_b   = (const float*)d_in[2];
  const float* gw     = (const float*)d_in[3];
  const float* gb     = (const float*)d_in[4];
  const float* lrw    = (const float*)d_in[5];
  const float* lrb    = (const float*)d_in[6];
  const float* ow     = (const float*)d_in[7];
  const float* ob     = (const float*)d_in[8];
  const float* mn_g   = (const float*)d_in[9];
  const float* mn_b   = (const float*)d_in[10];
  const float* w1     = (const float*)d_in[11];
  const float* b1     = (const float*)d_in[12];
  const float* w2     = (const float*)d_in[13];
  const float* b2     = (const float*)d_in[14];
  float* out = (float*)d_out;

  char* ws = (char*)d_ws;
  size_t off = 0;
  auto alloc = [&](size_t bytes) { char* p = ws + off; off += (bytes + 255) & ~255ULL; return p; };
  __bf16* h_bf   = (__bf16*)alloc(NT * C * 2);   // LN out (reused for LN2)
  __bf16* q_bf   = (__bf16*)alloc(NT * C * 2);   // q (reused as y)
  __bf16* k_bf   = (__bf16*)alloc(NT * C * 2);   // k (reused as MLP hidden)
  __bf16* v_bf   = (__bf16*)alloc(NT * C * 2);
  __bf16* u_bf   = (__bf16*)alloc(NT * C * 2);
  __bf16* ulr_bf = (__bf16*)alloc(NT * C * 2);
  __bf16* Kmat   = (__bf16*)alloc(Nb * T * T * 2);  // K (reused as score)
  __bf16* gw_bf  = (__bf16*)alloc(3 * C * C * 2);
  __bf16* ow_bf  = (__bf16*)alloc(C * C * 2);
  __bf16* w1_bf  = (__bf16*)alloc(C * C * 2);
  __bf16* w2_bf  = (__bf16*)alloc(C * C * 2);
  float*  lr     = (float*)alloc(NT * 4);
  (void)ws_size; (void)n_in; (void)in_sizes; (void)out_size;

  // weight conversions (fp32 -> bf16)
  cvt_kernel<<<dim3(1024), 256, 0, stream>>>(gw, gw_bf, 3 * C * C);
  cvt_kernel<<<dim3(512), 256, 0, stream>>>(ow, ow_bf, C * C);
  cvt_kernel<<<dim3(512), 256, 0, stream>>>(w1, w1_bf, C * C);
  cvt_kernel<<<dim3(512), 256, 0, stream>>>(w2, w2_bf, C * C);

  // LN1 + lr dot
  ln_kernel<true><<<dim3(NT), 256, 0, stream>>>(x, dn_g, dn_b, lrw, lrb, h_bf, lr);

  // qkv projections: h [NT,C] @ gates_w[:, sC:(s+1)C]
  __bf16* qkv[3] = {q_bf, k_bf, v_bf};
  for (int s = 0; s < 3; s++)
    gemm_kernel<false, true, false, false, false, true>
        <<<dim3(C / 128, NT / 128, 1), 256, 0, stream>>>(
            h_bf, C, 0, gw_bf + s * C, 3 * C, 0, gb + s * C,
            qkv[s], C, 0, nullptr, 0, (int)C);

  // normalize k rows
  knorm_kernel<<<dim3(NT), 256, 0, stream>>>(k_bf);

  // K = tril(k k^T, -1) per batch (upper blocks skipped)
  gemm_kernel<true, false, false, false, true, true>
      <<<dim3(T / 128, T / 128, Nb), 256, 0, stream>>>(
          k_bf, C, T * C, k_bf, C, T * C, nullptr,
          Kmat, T, T * T, nullptr, 0, (int)C);

  // blocked forward substitution: (I + K) u = v
  for (int step = 0; step < 16; step++)
    solve_step_kernel<<<dim3(C / 128, Nb), 256, 0, stream>>>(
        Kmat, u_bf, ulr_bf, v_bf, lr, step);

  // score = q k^T (no mask), reuse Kmat buffer
  gemm_kernel<true, false, false, false, false, true>
      <<<dim3(T / 128, T / 128, Nb), 256, 0, stream>>>(
          q_bf, C, T * C, k_bf, C, T * C, nullptr,
          Kmat, T, T * T, nullptr, 0, (int)C);

  // y = score @ (lr*u), reuse q_bf as y
  gemm_kernel<false, false, false, false, false, true>
      <<<dim3(C / 128, T / 128, Nb), 256, 0, stream>>>(
          Kmat, T, T * T, ulr_bf, C, T * C, nullptr,
          q_bf, C, T * C, nullptr, 0, (int)T);

  // x1 = x + y @ out_w + out_b  -> d_out (fp32)
  gemm_kernel<false, true, false, true, false, false>
      <<<dim3(C / 128, NT / 128, 1), 256, 0, stream>>>(
          q_bf, C, 0, ow_bf, C, 0, ob, out, C, 0, x, 0, (int)C);

  // LN2
  ln_kernel<false><<<dim3(NT), 256, 0, stream>>>(out, mn_g, mn_b, nullptr, nullptr, h_bf, nullptr);

  // a = relu(h2 @ w1 + b1), reuse k_bf
  gemm_kernel<false, true, true, false, false, true>
      <<<dim3(C / 128, NT / 128, 1), 256, 0, stream>>>(
          h_bf, C, 0, w1_bf, C, 0, b1, k_bf, C, 0, nullptr, 0, (int)C);

  // out = x1 + a @ w2 + b2 (in-place residual on d_out)
  gemm_kernel<false, true, false, true, false, false>
      <<<dim3(C / 128, NT / 128, 1), 256, 0, stream>>>(
          k_bf, C, 0, w2_bf, C, 0, b2, out, C, 0, out, 0, (int)C);
}
